// TransformerDecoder_50337016709840
// MI455X (gfx1250) — compile-verified, matrix operand order checked by reading, not run on hardware
//
#include <hip/hip_runtime.h>

// ---------------------------------------------------------------------------
// Transformer decoder (L=6,B=4,N=1024,D=1024,H=16,F=4096) for MI455X gfx1250.
// All matmuls in bf16 WMMA (v_wmma_f32_16x16x32_bf16, f32 accum); softmax/LN fp32.
// Compute-bound (~1 TFLOP vs ~30us of HBM traffic at 23.3TB/s) -> matrix pipe.
// ---------------------------------------------------------------------------

#define L_ 6
#define B_ 4
#define N_ 1024
#define D_ 1024
#define H_ 16
#define F_ 4096
#define HD_ 64

typedef __attribute__((ext_vector_type(16))) __bf16 v16bf;
typedef __attribute__((ext_vector_type(8)))  __bf16 v8bf;
typedef __attribute__((ext_vector_type(8)))  float  v8f;

static __device__ __forceinline__ __bf16 f2bf(float f) {
  return (__bf16)f;   // native v_cvt on gfx1250
}

static __device__ __forceinline__ v16bf cat8(v8bf lo, v8bf hi) {
  return __builtin_shufflevector(lo, hi, 0,1,2,3,4,5,6,7,8,9,10,11,12,13,14,15);
}

static __device__ __forceinline__ v8f v8f_zero() {
  v8f z;
#pragma unroll
  for (int i = 0; i < 8; ++i) z[i] = 0.0f;
  return z;
}

// ---------------------------------------------------------------------------
// GEMM: C[M,N] = act( (A[M,K] @ W[N,K]^T + bias) * scale ) (+ residual)
// Block: 256 thr = 8 waves, tile 64(M) x 256(N); wave tile 32x64; K-step 32.
// LDS staging in bf16 with row pad so fragment reads hit 16 distinct banks.
// RELU / RES are compile-time so the epilogue is branch-free.
// ---------------------------------------------------------------------------
template <bool RELU, bool RES>
__global__ __launch_bounds__(256)
void gemm_bias_kernel(const float* __restrict__ A, const float* __restrict__ W,
                      const float* __restrict__ bias, const float* __restrict__ R,
                      float* __restrict__ C, int M, int K, int N, float scale) {
  __shared__ __bf16 As[64 * 40];    // [row][k], stride 40 (pad 8)
  __shared__ __bf16 Ws[256 * 40];

  const int tid  = threadIdx.x;
  const int lane = tid & 31;
  const int wid  = tid >> 5;
  const int half = lane >> 4;       // 0: lanes 0-15, 1: lanes 16-31
  const int l16  = lane & 15;
  const int wm   = wid & 1;         // 2 wave rows
  const int wn   = wid >> 1;        // 4 wave cols
  const int mblk = blockIdx.x * 64;
  const int nblk = blockIdx.y * 256;
  const int koff = half * 8;        // ISA 16-bit A/B fragment K swizzle

  v8f acc[2][4];
#pragma unroll
  for (int i = 0; i < 2; ++i)
#pragma unroll
    for (int j = 0; j < 4; ++j) acc[i][j] = v8f_zero();

  const int r = tid >> 3;           // 0..31
  const int c = (tid & 7) * 4;      // 0..28

  for (int k0 = 0; k0 < K; k0 += 32) {
    // ---- cooperative staging (fp32 -> bf16) ----
#pragma unroll
    for (int p = 0; p < 2; ++p) {    // A: 64 rows
      const int row = r + p * 32;
      const float4 f = *(const float4*)(A + (size_t)(mblk + row) * K + k0 + c);
      __bf16* d = &As[row * 40 + c];
      d[0] = f2bf(f.x); d[1] = f2bf(f.y); d[2] = f2bf(f.z); d[3] = f2bf(f.w);
    }
#pragma unroll
    for (int p = 0; p < 8; ++p) {    // W: 256 rows
      const int row = r + p * 32;
      const float4 f = *(const float4*)(W + (size_t)(nblk + row) * K + k0 + c);
      __bf16* d = &Ws[row * 40 + c];
      d[0] = f2bf(f.x); d[1] = f2bf(f.y); d[2] = f2bf(f.z); d[3] = f2bf(f.w);
    }
    // prefetch next K-tile into cache while this one is consumed
    if (k0 + 32 < K) {
      __builtin_prefetch(A + (size_t)(mblk + r) * K + k0 + 32 + c, 0, 1);
      __builtin_prefetch(W + (size_t)(nblk + r) * K + k0 + 32 + c, 0, 1);
      __builtin_prefetch(W + (size_t)(nblk + r + 128) * K + k0 + 32 + c, 0, 1);
    }
    __syncthreads();

    // ---- fragment build + 8 WMMAs ----
    v16bf a[2], b[4];
#pragma unroll
    for (int ms = 0; ms < 2; ++ms) {
      const __bf16* p = &As[(wm * 32 + ms * 16 + l16) * 40 + koff];
      a[ms] = cat8(*(const v8bf*)p, *(const v8bf*)(p + 16));
    }
#pragma unroll
    for (int ns = 0; ns < 4; ++ns) {
      const __bf16* p = &Ws[(wn * 64 + ns * 16 + l16) * 40 + koff];
      b[ns] = cat8(*(const v8bf*)p, *(const v8bf*)(p + 16));
    }
#pragma unroll
    for (int ms = 0; ms < 2; ++ms)
#pragma unroll
      for (int ns = 0; ns < 4; ++ns)
        acc[ms][ns] = __builtin_amdgcn_wmma_f32_16x16x32_bf16(
            false, a[ms], false, b[ns], (short)0, acc[ms][ns], false, false);
    __syncthreads();
  }

  // ---- epilogue: bias, scale, relu, residual (branch-free) ----
#pragma unroll
  for (int ms = 0; ms < 2; ++ms) {
#pragma unroll
    for (int ns = 0; ns < 4; ++ns) {
      const int col = nblk + wn * 64 + ns * 16 + l16;
      const float bv = bias[col];
#pragma unroll
      for (int rr = 0; rr < 8; ++rr) {
        const int row = mblk + wm * 32 + ms * 16 + rr + half * 8;
        float v = (acc[ms][ns][rr] + bv) * scale;
        if (RELU) v = fmaxf(v, 0.0f);
        if (RES) v += R[(size_t)row * N + col];
        C[(size_t)row * N + col] = v;
      }
    }
  }
}

// ---------------------------------------------------------------------------
// Flash attention: O = softmax(Q K^T + mask) V, per (b,h).  Q pre-scaled.
// Block: 256 thr = 8 waves; slab of 128 rows (16/wave); j-step 32 cols.
// K staged row-major, V staged transposed (so PV B-frags are contiguous).
// ---------------------------------------------------------------------------
__global__ __launch_bounds__(256)
void attention_kernel(const float* __restrict__ Q, const float* __restrict__ Kk,
                      const float* __restrict__ V, const float* __restrict__ mask,
                      float* __restrict__ O) {
  __shared__ __bf16 Ks[32 * 72];       // [j'][d]  (pad 8 -> 144B rows)
  __shared__ __bf16 Vs[64 * 40];       // [d][j']  transposed
  __shared__ __bf16 Ps[8][16 * 40];    // per-wave P reswizzle buffer

  const int tid  = threadIdx.x;
  const int lane = tid & 31;
  const int wid  = tid >> 5;
  const int half = lane >> 4;
  const int l16  = lane & 15;
  const int b    = blockIdx.x / H_;
  const int h    = blockIdx.x % H_;
  const int hoff = h * HD_;
  const int ibase = blockIdx.y * 128 + wid * 16;
  const int koff  = half * 8;

  // Q fragments (held across the whole j loop), K dim = HD = 64 -> 2 frags
  v16bf aq[2];
  {
    const float* qr = Q + ((size_t)b * N_ + ibase + l16) * D_ + hoff;
#pragma unroll
    for (int dh = 0; dh < 2; ++dh) {
      const float4 f0 = *(const float4*)(qr + dh * 32 + koff);
      const float4 f1 = *(const float4*)(qr + dh * 32 + koff + 4);
      const float4 f2 = *(const float4*)(qr + dh * 32 + koff + 16);
      const float4 f3 = *(const float4*)(qr + dh * 32 + koff + 20);
      v16bf t;
      t[0] = f2bf(f0.x); t[1]  = f2bf(f0.y); t[2]  = f2bf(f0.z); t[3]  = f2bf(f0.w);
      t[4] = f2bf(f1.x); t[5]  = f2bf(f1.y); t[6]  = f2bf(f1.z); t[7]  = f2bf(f1.w);
      t[8] = f2bf(f2.x); t[9]  = f2bf(f2.y); t[10] = f2bf(f2.z); t[11] = f2bf(f2.w);
      t[12] = f2bf(f3.x); t[13] = f2bf(f3.y); t[14] = f2bf(f3.z); t[15] = f2bf(f3.w);
      aq[dh] = t;
    }
  }

  float mrow[8], lrow[8];
  v8f oacc[4];
#pragma unroll
  for (int rr = 0; rr < 8; ++rr) { mrow[rr] = -1e30f; lrow[rr] = 0.0f; }
#pragma unroll
  for (int t = 0; t < 4; ++t) oacc[t] = v8f_zero();

  const int sr = tid >> 4;           // 0..15
  const int sc = (tid & 15) * 4;     // 0..60

  for (int j0 = 0; j0 < N_; j0 += 32) {
    // ---- stage K (row major) and V (transposed) as bf16 ----
#pragma unroll
    for (int p = 0; p < 2; ++p) {
      const int row = sr + p * 16;     // 0..31
      const float4 f = *(const float4*)(Kk + ((size_t)b * N_ + j0 + row) * D_ + hoff + sc);
      __bf16* d = &Ks[row * 72 + sc];
      d[0] = f2bf(f.x); d[1] = f2bf(f.y); d[2] = f2bf(f.z); d[3] = f2bf(f.w);
      const float4 g = *(const float4*)(V + ((size_t)b * N_ + j0 + row) * D_ + hoff + sc);
      Vs[(sc + 0) * 40 + row] = f2bf(g.x);
      Vs[(sc + 1) * 40 + row] = f2bf(g.y);
      Vs[(sc + 2) * 40 + row] = f2bf(g.z);
      Vs[(sc + 3) * 40 + row] = f2bf(g.w);
    }
    if (j0 + 32 < N_) {
      __builtin_prefetch(Kk + ((size_t)b * N_ + j0 + 32 + sr) * D_ + hoff + sc, 0, 1);
      __builtin_prefetch(V  + ((size_t)b * N_ + j0 + 32 + sr) * D_ + hoff + sc, 0, 1);
    }
    __syncthreads();

    // ---- S = Q K^T : 16 rows x 32 cols (2 subtiles x 2 K-halves) ----
    v8f s[2];
#pragma unroll
    for (int js = 0; js < 2; ++js) {
      const __bf16* p = &Ks[(js * 16 + l16) * 72];
      const v16bf bk0 = cat8(*(const v8bf*)(p + koff),      *(const v8bf*)(p + koff + 16));
      const v16bf bk1 = cat8(*(const v8bf*)(p + 32 + koff), *(const v8bf*)(p + 32 + koff + 16));
      v8f z = v8f_zero();
      z = __builtin_amdgcn_wmma_f32_16x16x32_bf16(false, aq[0], false, bk0, (short)0, z, false, false);
      z = __builtin_amdgcn_wmma_f32_16x16x32_bf16(false, aq[1], false, bk1, (short)0, z, false, false);
      s[js] = z;
    }

    // ---- + mask ----
#pragma unroll
    for (int js = 0; js < 2; ++js)
#pragma unroll
      for (int rr = 0; rr < 8; ++rr) {
        const int i = ibase + rr + half * 8;
        const int j = j0 + js * 16 + l16;
        s[js][rr] += mask[(size_t)i * N_ + j];
      }

    // ---- online softmax (row lives in one 16-lane half of the wave) ----
#pragma unroll
    for (int rr = 0; rr < 8; ++rr) {
      float mx = fmaxf(s[0][rr], s[1][rr]);
#pragma unroll
      for (int d = 1; d < 16; d <<= 1) mx = fmaxf(mx, __shfl_xor(mx, d, 32));
      const float mnew = fmaxf(mrow[rr], mx);
      const float corr = __expf(mrow[rr] - mnew);
      const float p0 = __expf(s[0][rr] - mnew);
      const float p1 = __expf(s[1][rr] - mnew);
      float sum = p0 + p1;
#pragma unroll
      for (int d = 1; d < 16; d <<= 1) sum += __shfl_xor(sum, d, 32);
      lrow[rr] = lrow[rr] * corr + sum;
      mrow[rr] = mnew;
      s[0][rr] = p0; s[1][rr] = p1;
#pragma unroll
      for (int t = 0; t < 4; ++t) oacc[t][rr] *= corr;
    }

    // ---- reswizzle P (C-layout f32 -> A-fragment bf16) via wave-private LDS
    // (same-wave LDS is in-order, no cross-wave sync needed) ----
    __bf16* pp = &Ps[wid][0];
#pragma unroll
    for (int js = 0; js < 2; ++js)
#pragma unroll
      for (int rr = 0; rr < 8; ++rr)
        pp[(rr + half * 8) * 40 + js * 16 + l16] = f2bf(s[js][rr]);
    const __bf16* pr = pp + l16 * 40 + koff;
    const v16bf ap = cat8(*(const v8bf*)pr, *(const v8bf*)(pr + 16));

    // ---- O += P V : 4 d-subtiles ----
#pragma unroll
    for (int t = 0; t < 4; ++t) {
      const __bf16* q = &Vs[(t * 16 + l16) * 40 + koff];
      const v16bf bv = cat8(*(const v8bf*)q, *(const v8bf*)(q + 16));
      oacc[t] = __builtin_amdgcn_wmma_f32_16x16x32_bf16(false, ap, false, bv, (short)0, oacc[t], false, false);
    }
    __syncthreads();
  }

  // ---- normalize and write O ----
#pragma unroll
  for (int t = 0; t < 4; ++t)
#pragma unroll
    for (int rr = 0; rr < 8; ++rr) {
      const int i = ibase + rr + half * 8;
      O[((size_t)b * N_ + i) * D_ + hoff + t * 16 + l16] = oacc[t][rr] / lrow[rr];
    }
}

// ---------------------------------------------------------------------------
// LayerNorm: one block per row of D=1024, 256 threads x 4 elems.
// ---------------------------------------------------------------------------
__global__ __launch_bounds__(256)
void layernorm_kernel(const float* __restrict__ X, const float* __restrict__ g,
                      const float* __restrict__ be, float* __restrict__ Y) {
  __shared__ float red[256];
  const int row = blockIdx.x;
  const int tid = threadIdx.x;
  const float4 v = *(const float4*)(X + (size_t)row * D_ + tid * 4);

  red[tid] = v.x + v.y + v.z + v.w;
  __syncthreads();
  for (int off = 128; off > 0; off >>= 1) {
    if (tid < off) red[tid] += red[tid + off];
    __syncthreads();
  }
  const float mean = red[0] * (1.0f / D_);
  __syncthreads();

  const float d0 = v.x - mean, d1 = v.y - mean, d2 = v.z - mean, d3 = v.w - mean;
  red[tid] = d0 * d0 + d1 * d1 + d2 * d2 + d3 * d3;
  __syncthreads();
  for (int off = 128; off > 0; off >>= 1) {
    if (tid < off) red[tid] += red[tid + off];
    __syncthreads();
  }
  const float rs = rsqrtf(red[0] * (1.0f / D_) + 1e-5f);

  const float4 gg = *(const float4*)(g + tid * 4);
  const float4 bb = *(const float4*)(be + tid * 4);
  float4 o;
  o.x = d0 * rs * gg.x + bb.x;
  o.y = d1 * rs * gg.y + bb.y;
  o.z = d2 * rs * gg.z + bb.z;
  o.w = d3 * rs * gg.w + bb.w;
  *(float4*)(Y + (size_t)row * D_ + tid * 4) = o;
}

// ---------------------------------------------------------------------------
// Host orchestration
// ---------------------------------------------------------------------------
extern "C" void kernel_launch(void* const* d_in, const int* in_sizes, int n_in,
                              void* d_out, int out_size, void* d_ws, size_t ws_size,
                              hipStream_t stream) {
  (void)in_sizes; (void)n_in; (void)out_size; (void)ws_size;

  const float* feats = (const float*)d_in[0];
  const float* enc   = (const float*)d_in[1];
  const float* trg   = (const float*)d_in[2];
  const float* src   = (const float*)d_in[3];
  const float* wq1 = (const float*)d_in[4],  * bq1 = (const float*)d_in[5];
  const float* wk1 = (const float*)d_in[6],  * bk1 = (const float*)d_in[7];
  const float* wv1 = (const float*)d_in[8],  * bv1 = (const float*)d_in[9];
  const float* wo1 = (const float*)d_in[10], * bo1 = (const float*)d_in[11];
  const float* wq2 = (const float*)d_in[12], * bq2 = (const float*)d_in[13];
  const float* wk2 = (const float*)d_in[14], * bk2 = (const float*)d_in[15];
  const float* wv2 = (const float*)d_in[16], * bv2 = (const float*)d_in[17];
  const float* wo2 = (const float*)d_in[18], * bo2 = (const float*)d_in[19];
  const float* wf1 = (const float*)d_in[20], * bf1 = (const float*)d_in[21];
  const float* wf2 = (const float*)d_in[22], * bf2 = (const float*)d_in[23];
  const float* g1 = (const float*)d_in[24], * be1 = (const float*)d_in[25];
  const float* g2 = (const float*)d_in[26], * be2 = (const float*)d_in[27];
  const float* g3 = (const float*)d_in[28], * be3 = (const float*)d_in[29];

  float* out = (float*)d_out;
  float* ws  = (float*)d_ws;
  const size_t BND = (size_t)B_ * N_ * D_;
  float* q  = ws;
  float* k  = q + BND;
  float* v  = k + BND;
  float* o  = v + BND;
  float* t  = o + BND;
  float* y  = t + BND;
  float* hb = y + BND;                 // [B,N,F]

  const int M = B_ * N_;               // 4096
  const dim3 blk(256);
  const dim3 gD(M / 64, D_ / 256);     // GEMM -> D cols
  const dim3 gF(M / 64, F_ / 256);     // GEMM -> F cols
  const dim3 gA(B_ * H_, N_ / 128);    // attention
  const dim3 gLN(M);                   // layernorm
  const float qs = 0.125f;             // HD^-0.5

  const float* x = enc;
  for (int l = 0; l < L_; ++l) {
    const float* feat = feats + (size_t)l * BND;
    const size_t wD = (size_t)l * D_ * D_, bD = (size_t)l * D_;
    const size_t wF = (size_t)l * F_ * D_, bF = (size_t)l * F_;

    // ---- attn1: cross-attention(feat -> enc), src_mask ----
    gemm_bias_kernel<false,false><<<gD, blk, 0, stream>>>(feat, wq1 + wD, bq1 + bD, nullptr, q, M, D_, D_, qs);
    gemm_bias_kernel<false,false><<<gD, blk, 0, stream>>>(enc,  wk1 + wD, bk1 + bD, nullptr, k, M, D_, D_, 1.0f);
    gemm_bias_kernel<false,false><<<gD, blk, 0, stream>>>(enc,  wv1 + wD, bv1 + bD, nullptr, v, M, D_, D_, 1.0f);
    attention_kernel<<<gA, blk, 0, stream>>>(q, k, v, src, o);
    gemm_bias_kernel<false,true><<<gD, blk, 0, stream>>>(o, wo1 + wD, bo1 + bD, feat, t, M, D_, D_, 1.0f);
    layernorm_kernel<<<gLN, blk, 0, stream>>>(t, g1 + bD, be1 + bD, y);

    // ---- attn2: self-attention(y -> prev decoder out x), trg_mask ----
    gemm_bias_kernel<false,false><<<gD, blk, 0, stream>>>(y, wq2 + wD, bq2 + bD, nullptr, q, M, D_, D_, qs);
    gemm_bias_kernel<false,false><<<gD, blk, 0, stream>>>(x, wk2 + wD, bk2 + bD, nullptr, k, M, D_, D_, 1.0f);
    gemm_bias_kernel<false,false><<<gD, blk, 0, stream>>>(x, wv2 + wD, bv2 + bD, nullptr, v, M, D_, D_, 1.0f);
    attention_kernel<<<gA, blk, 0, stream>>>(q, k, v, trg, o);
    gemm_bias_kernel<false,true><<<gD, blk, 0, stream>>>(o, wo2 + wD, bo2 + bD, y, t, M, D_, D_, 1.0f);
    layernorm_kernel<<<gLN, blk, 0, stream>>>(t, g2 + bD, be2 + bD, o);   // o = y2

    // ---- FFN ----
    gemm_bias_kernel<true,false><<<gF, blk, 0, stream>>>(o,  wf1 + wF, bf1 + bF, nullptr, hb, M, D_, F_, 1.0f);
    gemm_bias_kernel<false,true><<<gD, blk, 0, stream>>>(hb, wf2 + wF, bf2 + bD, o, t, M, F_, D_, 1.0f);
    layernorm_kernel<<<gLN, blk, 0, stream>>>(t, g3 + bD, be3 + bD, out + (size_t)l * BND);

    x = out + (size_t)l * BND;
  }
}
